// Model_ConvLSTM_695784702586
// MI455X (gfx1250) — matrix operand order repaired and checked by reading, compiled, not run on hardware
//
#include <hip/hip_runtime.h>
#include <hip/hip_bf16.h>
#include <stdint.h>

typedef __attribute__((ext_vector_type(16))) _Float16 v16h;
typedef __attribute__((ext_vector_type(8)))  float    v8f;

#define NFT 64
#define BB 2
#define TT 6
#define FUTT 6
#define HH 64
#define WWI 64
#define HW (HH*WWI)            // 4096
#define NTOT (BB*HW)           // 8192
#define MMC (4*NFT)            // 256

// ---------------- zero fill (states) ----------------
__global__ void fill_zero_f4(float4* p, long n4) {
  long i = (long)blockIdx.x * blockDim.x + threadIdx.x;
  if (i < n4) p[i] = make_float4(0.f, 0.f, 0.f, 0.f);
}

// ---------------- weight f32 -> f16 with K zero-pad ----------------
__global__ void convert_w(_Float16* __restrict__ dst, const float* __restrict__ src,
                          int K, int Kpad) {
  int i = blockIdx.x * blockDim.x + threadIdx.x;
  if (i >= MMC * Kpad) return;
  int co = i / Kpad, k = i - co * Kpad;
  dst[i] = (k < K) ? (_Float16)src[co * K + k] : (_Float16)0.f;
}

// ---------------- im2col, transposed patch matrix BmatT[n][k] (f16) ----------------
// channel c < CA comes from srcA (conv input), c >= CA from srcB (hidden state)
__global__ void im2col(_Float16* __restrict__ Bm,
                       const float* __restrict__ srcA, int CA, int strideA,
                       const float* __restrict__ srcB, int CB, int strideB,
                       int Kpad) {
  int C = CA + CB;
  int total = BB * C * HW;
  int i = blockIdx.x * blockDim.x + threadIdx.x;
  if (i >= total) return;
  int b = i / (C * HW);
  int r = i - b * C * HW;
  int c = r / HW;
  int p = r - c * HW;
  int y = p >> 6, x = p & 63;
  const float* s = (c < CA) ? (srcA + (size_t)b * strideA + (size_t)c * HW)
                            : (srcB + (size_t)b * strideB + (size_t)(c - CA) * HW);
  size_t n = (size_t)b * HW + p;
  _Float16* d = Bm + n * (size_t)Kpad + (size_t)c * 9;
  #pragma unroll
  for (int dy = -1; dy <= 1; ++dy) {
    #pragma unroll
    for (int dx = -1; dx <= 1; ++dx) {
      int yy = y + dy, xx = x + dx;
      float v = (yy >= 0 && yy < HH && xx >= 0 && xx < WWI) ? s[yy * WWI + xx] : 0.f;
      d[(dy + 1) * 3 + (dx + 1)] = (_Float16)v;
    }
  }
}

// zero the K..Kpad pad columns of BmatT (only needed for cell 0)
__global__ void pad_rows(_Float16* __restrict__ Bm, int K, int Kpad) {
  int pad = Kpad - K;
  int i = blockIdx.x * blockDim.x + threadIdx.x;
  if (i >= NTOT * pad) return;
  int n = i / pad, j = i - n * pad;
  Bm[(size_t)n * Kpad + K + j] = (_Float16)0.f;
}

// ---------------- implicit-GEMM conv core: Z[m][n] = sum_k A[m][k]*B[n][k] ----------------
// A: weights [MMC][Kpad] f16 (row-major, k-contiguous)
// B: BmatT   [NTOT][Kpad] f16 (row-major, k-contiguous)
// Workgroup = 256 threads (8 waves), block tile = 128(M) x 64(N), K chunked by 32.
// Each wave owns a 32x32 sub-tile = 2x2 WMMA tiles (4 accumulators, shared fragments).
__launch_bounds__(256)
__global__ void gemm_conv(float* __restrict__ Z,
                          const _Float16* __restrict__ A,
                          const _Float16* __restrict__ Bm,
                          int Kpad) {
  __shared__ _Float16 As[128 * 32];   // [m][k]  8 KB
  __shared__ _Float16 Bs[64 * 32];    // [n][k]  4 KB
  int tid  = threadIdx.x;
  int n0   = blockIdx.x * 64;
  int m0   = blockIdx.y * 128;
  int w    = tid >> 5, lane = tid & 31;
  int bm   = (w & 3) * 32;   // wave M base within block (0,32,64,96)
  int bn   = (w >> 2) * 32;  // wave N base within block (0,32)
  int ml   = lane & 15;      // row/col within 16
  int kg   = lane >> 4;      // lane half-group
  int row  = tid >> 2;       // staging: 4 threads/row
  int seg  = tid & 3;        // 8 halves (16B) per thread
  v8f a00 = {}, a01 = {}, a10 = {}, a11 = {};
  int chunks = Kpad >> 5;
  for (int kc = 0; kc < chunks; ++kc) {
    int k0 = kc << 5;
    uint4 va0 = *(const uint4*)(A  + (size_t)(m0 + row)      * Kpad + k0 + seg * 8);
    uint4 va1 = *(const uint4*)(A  + (size_t)(m0 + row + 64) * Kpad + k0 + seg * 8);
    uint4 vb  = *(const uint4*)(Bm + (size_t)(n0 + row)      * Kpad + k0 + seg * 8);
    __syncthreads();                       // prior chunk fully consumed
    *(uint4*)&As[row * 32 + seg * 8]        = va0;
    *(uint4*)&As[(row + 64) * 32 + seg * 8] = va1;
    *(uint4*)&Bs[row * 32 + seg * 8]        = vb;
    __syncthreads();                       // chunk staged
    // A fragments (16x32): lane m=ml, two contiguous 8-half runs at kg*8 and 16+kg*8
    union { uint4 u[2]; v16h v; } fa0, fa1, fb0, fb1;
    const _Float16* ap0 = &As[(bm + ml) * 32 + kg * 8];
    fa0.u[0] = *(const uint4*)ap0;
    fa0.u[1] = *(const uint4*)(ap0 + 16);
    const _Float16* ap1 = &As[(bm + 16 + ml) * 32 + kg * 8];
    fa1.u[0] = *(const uint4*)ap1;
    fa1.u[1] = *(const uint4*)(ap1 + 16);
    // B fragments (32x16): lane n=ml, two contiguous 8-half runs at kg*8 and 16+kg*8
    const _Float16* bp0 = &Bs[(bn + ml) * 32 + kg * 8];
    fb0.u[0] = *(const uint4*)bp0;
    fb0.u[1] = *(const uint4*)(bp0 + 16);
    const _Float16* bp1 = &Bs[(bn + 16 + ml) * 32 + kg * 8];
    fb1.u[0] = *(const uint4*)bp1;
    fb1.u[1] = *(const uint4*)(bp1 + 16);
    a00 = __builtin_amdgcn_wmma_f32_16x16x32_f16(false, fa0.v, false, fb0.v,
                                                 (short)0, a00, false, false);
    a01 = __builtin_amdgcn_wmma_f32_16x16x32_f16(false, fa0.v, false, fb1.v,
                                                 (short)0, a01, false, false);
    a10 = __builtin_amdgcn_wmma_f32_16x16x32_f16(false, fa1.v, false, fb0.v,
                                                 (short)0, a10, false, false);
    a11 = __builtin_amdgcn_wmma_f32_16x16x32_f16(false, fa1.v, false, fb1.v,
                                                 (short)0, a11, false, false);
  }
  // C/D layout: VGPR v -> M = v + 8*kg ; N = lane%16
  int col0 = n0 + bn + ml;
  int mr0  = m0 + bm + 8 * kg;
  #pragma unroll
  for (int v = 0; v < 8; ++v) {
    Z[(size_t)(mr0 + v)      * NTOT + col0]      = a00[v];
    Z[(size_t)(mr0 + v)      * NTOT + col0 + 16] = a01[v];
    Z[(size_t)(mr0 + 16 + v) * NTOT + col0]      = a10[v];
    Z[(size_t)(mr0 + 16 + v) * NTOT + col0 + 16] = a11[v];
  }
}

// ---------------- LSTM gates (+ optional ReLU output write) ----------------
__device__ __forceinline__ float sigmoidf_(float x) { return 1.f / (1.f + __expf(-x)); }

__global__ void gate_kernel(const float* __restrict__ Z, const float* __restrict__ bias,
                            const float* __restrict__ cprev, float* __restrict__ hout,
                            float* __restrict__ cout, float* __restrict__ outp, int tslot) {
  int i = blockIdx.x * blockDim.x + threadIdx.x;
  if (i >= BB * NFT * HW) return;
  int b  = i / (NFT * HW);
  int r  = i - b * NFT * HW;
  int ch = r / HW;
  int p  = r - ch * HW;
  int n  = b * HW + p;
  float zi = Z[(size_t)(0 * NFT + ch) * NTOT + n] + bias[0 * NFT + ch];
  float zf = Z[(size_t)(1 * NFT + ch) * NTOT + n] + bias[1 * NFT + ch];
  float zo = Z[(size_t)(2 * NFT + ch) * NTOT + n] + bias[2 * NFT + ch];
  float zg = Z[(size_t)(3 * NFT + ch) * NTOT + n] + bias[3 * NFT + ch];
  float c2 = sigmoidf_(zf) * cprev[i] + sigmoidf_(zi) * tanhf(zg);
  float h2 = sigmoidf_(zo) * tanhf(c2);
  cout[i] = c2;
  hout[i] = h2;
  if (outp) outp[((size_t)(b * NFT + ch) * FUTT + tslot) * HW + p] = fmaxf(h2, 0.f);
}

// ---------------- host orchestration ----------------
extern "C" void kernel_launch(void* const* d_in, const int* in_sizes, int n_in,
                              void* d_out, int out_size, void* d_ws, size_t ws_size,
                              hipStream_t stream) {
  const float* x = (const float*)d_in[0];
  const float* Wp[12]; const float* bp[12];
  for (int i = 0; i < 12; ++i) Wp[i] = (const float*)d_in[1 + i];
  for (int i = 0; i < 12; ++i) bp[i] = (const float*)d_in[13 + i];
  float* out = (float*)d_out;

  int K[12], Kp[12];
  K[0] = 65 * 9;  Kp[0] = ((K[0] + 31) / 32) * 32;       // 585 -> 608
  for (int i = 1; i < 12; ++i) { K[i] = 128 * 9; Kp[i] = 1152; }

  char* ws = (char*)d_ws;
  size_t off = 0;
  _Float16* Wf[12];
  for (int i = 0; i < 12; ++i) { Wf[i] = (_Float16*)(ws + off); off += (size_t)MMC * Kp[i] * 2; }
  off = (off + 255) & ~(size_t)255;
  _Float16* Bm = (_Float16*)(ws + off); off += (size_t)NTOT * 1152 * 2;
  float* Z = (float*)(ws + off);        off += (size_t)MMC * NTOT * 4;
  float* states = (float*)(ws + off);
  const size_t SB = (size_t)BB * NFT * HW;   // 524288 elems / state
  float* h1 = states;     float* c1 = h1 + SB;
  float* h2 = c1 + SB;    float* c2 = h2 + SB;
  float* h3 = c2 + SB;    float* h4 = h3 + SB;
  float* h5 = h4 + SB;    float* h6 = h5 + SB;
  float* dh[6]; float* dc[6];
  float* pp = h6 + SB;
  for (int k = 0; k < 6; ++k) { dh[k] = pp; pp += SB; }
  for (int k = 0; k < 6; ++k) { dc[k] = pp; pp += SB; }
  float* ctrash = pp;

  // zero all 21 state buffers every launch (deterministic under graph replay)
  long n4 = (long)(21 * SB) / 4;
  fill_zero_f4<<<(int)((n4 + 255) / 256), 256, 0, stream>>>((float4*)states, n4);

  for (int i = 0; i < 12; ++i)
    convert_w<<<(MMC * Kp[i] + 255) / 256, 256, 0, stream>>>(Wf[i], Wp[i], K[i], Kp[i]);

  auto conv = [&](int cell, const float* sa, int CA, int sA, const float* sb) {
    int tot = BB * (CA + NFT) * HW;
    im2col<<<(tot + 255) / 256, 256, 0, stream>>>(Bm, sa, CA, sA, sb, NFT, NFT * HW, Kp[cell]);
    if (Kp[cell] > K[cell]) {
      int pt = NTOT * (Kp[cell] - K[cell]);
      pad_rows<<<(pt + 255) / 256, 256, 0, stream>>>(Bm, K[cell], Kp[cell]);
    }
    dim3 g(NTOT / 64, MMC / 128);
    gemm_conv<<<g, 256, 0, stream>>>(Z, Wf[cell], Bm, Kp[cell]);
  };
  auto gate = [&](int cell, const float* cp, float* ho, float* co, float* op, int t) {
    int tot = BB * NFT * HW;
    gate_kernel<<<(tot + 255) / 256, 256, 0, stream>>>(Z, bp[cell], cp, ho, co, op, t);
  };

  // ---- encoder over T steps (cells 2..5 use c := h, discard c2) ----
  for (int t = 0; t < TT; ++t) {
    conv(0, x + (size_t)t * HW, 1, TT * HW, h1);  gate(0, c1, h1, c1, nullptr, 0);
    conv(1, h1, NFT, NFT * HW, h2);               gate(1, c2, h2, c2, nullptr, 0);
    conv(2, h2, NFT, NFT * HW, h3);               gate(2, h3, h3, ctrash, nullptr, 0);
    conv(3, h3, NFT, NFT * HW, h4);               gate(3, h4, h4, ctrash, nullptr, 0);
    conv(4, h4, NFT, NFT * HW, h5);               gate(4, h5, h5, ctrash, nullptr, 0);
    conv(5, h5, NFT, NFT * HW, h6);               gate(5, h6, h6, ctrash, nullptr, 0);
  }
  // ---- decoder: constant input h6, proper (h,c); last cell writes ReLU output ----
  for (int t = 0; t < FUTT; ++t) {
    for (int k = 0; k < 6; ++k) {
      const float* inp = (k == 0) ? h6 : dh[k - 1];
      conv(6 + k, inp, NFT, NFT * HW, dh[k]);
      gate(6 + k, dc[k], dh[k], dc[k], (k == 5) ? out : nullptr, t);
    }
  }
  (void)in_sizes; (void)n_in; (void)out_size; (void)ws_size;
}
